// SimpleRNN_65034394796939
// MI455X (gfx1250) — matrix operand order repaired and checked by reading, not compiled, not run on hardware
//
#include <hip/hip_runtime.h>

// ---------------------------------------------------------------------------
// SimpleRNN on gfx1250 (MI455X):
//   x = emb[tokens]                          [64,512,512]
//   h1 = scan(tanh(x@W1 + b1 + h@U1))        seq out   [64,512,256]
//   h2 = scan(tanh(h1@W2 + b2 + h@U2))       last out  [64,256]
//   out = softmax(h2@Wd + bd)                [64,2]
// bf16 WMMA (v_wmma_f32_16x16x32_bf16) everywhere; fp32 accumulation.
// Scan uses CDNA5 split workgroup barrier (s_barrier_signal/-wait) with the
// next-step xw loads software-pipelined into the signal->wait window.
// ---------------------------------------------------------------------------

typedef __attribute__((ext_vector_type(16))) __bf16 v16bf;
typedef __attribute__((ext_vector_type(8)))  float  v8f;

union Frag16 {
    v16bf          v;
    unsigned short u16[16];
    unsigned int   u32[8];
};

__device__ __forceinline__ unsigned short f2bf_bits(float f) {
    union { float f; unsigned int u; } x;
    x.f = f;
    // round-to-nearest-even bf16 conversion
    return (unsigned short)((x.u + 0x7FFFu + ((x.u >> 16) & 1u)) >> 16);
}

__device__ __forceinline__ float fast_tanh(float x) {
    float ax = fabsf(x);
    float e  = __expf(-2.0f * ax);          // v_exp_f32
    float t  = (1.0f - e) / (1.0f + e);
    return x < 0.0f ? -t : t;
}

// CDNA5 split workgroup barrier (barrier id -1 = workgroup barrier).
// SIGNAL: retire our LDS traffic (dscnt) then arrive. WAIT: block until all
// waves arrived. "memory" clobbers pin LDS access ordering around each half.
#define WG_BARRIER_SIGNAL() \
    asm volatile("s_wait_dscnt 0x0\n\ts_barrier_signal -1" ::: "memory")
#define WG_BARRIER_WAIT() \
    asm volatile("s_barrier_wait -1" ::: "memory")

// ---------------------------------------------------------------------------
// Projection GEMM:  out[M,256] = A[M,K] @ Bmat[K,256] + bias
//   GATHER=true : A row m = emb[tokens[b][s]]  (m = s*64 + b), K=512, fp32->bf16
//   GATHER=false: A from bf16 buffer (scan-1 sequence output), K=256
// Workgroup: 256 threads (8 waves), 64 rows x 256 cols per block.
// Wave w owns n-tiles {2w, 2w+1}; K staged in chunks of 64 through LDS.
// ---------------------------------------------------------------------------
template <bool GATHER>
__global__ __launch_bounds__(256) void proj_kernel(
    const int* __restrict__ tokens, const float* __restrict__ emb,
    const unsigned short* __restrict__ Abf, const float* __restrict__ Bmat,
    const float* __restrict__ bias, float* __restrict__ out, int K)
{
    __shared__ unsigned short As[64][64];        //  8 KB: A chunk, bf16 row-major
    __shared__ unsigned short Bs[2][16][512];    // 32 KB: B chunk, fragment-major

    const int  tid  = threadIdx.x;
    const int  lane = tid & 31;
    const int  wave = tid >> 5;
    const int  half = lane >> 4;
    const int  l15  = lane & 15;
    const long b0   = (long)blockIdx.x * 64;

    v8f acc[4][2] = {};

    const int nchunk = K >> 6;
    for (int ch = 0; ch < nchunk; ++ch) {
        const int ko = ch << 6;

        // warm L2 for the next weight chunk (global_prefetch_b8)
        if (ch + 1 < nchunk) {
            int kpf = ((ch + 1) << 6) + (tid >> 2);      // 64 k-rows ahead
            __builtin_prefetch(&Bmat[(size_t)kpf * 256 + (tid & 3) * 64], 0, 3);
        }

        __syncthreads();

        // ---- stage A chunk: 64 rows x 64 k (4096 elems, 16/thread) ----
        for (int i = 0; i < 16; ++i) {
            int idx = tid + (i << 8);
            int row = idx >> 6, kk = idx & 63;
            if (GATHER) {
                long m = b0 + row;                 // m = s*64 + b
                int  s = (int)(m >> 6), b = (int)(m & 63);
                int  tok = tokens[b * 512 + s];
                As[row][kk] = f2bf_bits(emb[(size_t)tok * 512 + ko + kk]);
            } else {
                As[row][kk] = Abf[(size_t)(b0 + row) * 256 + ko + kk];
            }
        }
        // ---- stage B chunk as WMMA fragments (16384 elems, 64/thread) ----
        for (int i = 0; i < 64; ++i) {
            int f  = tid + (i << 8);
            int e  = f & 15, ln = (f >> 4) & 31, nt = (f >> 9) & 15, kt = f >> 13;
            int vv = e >> 1, odd = e & 1, hf = ln >> 4;
            int n  = nt * 16 + (ln & 15);
            int k  = ko + kt * 32 + hf * 16 + vv * 2 + odd;
            Bs[kt][nt][ln * 16 + e] = f2bf_bits(Bmat[(size_t)k * 256 + n]);
        }
        __syncthreads();

        // ---- compute: 2 k-tiles x 4 m-tiles x 2 n-tiles ----
        for (int kt = 0; kt < 2; ++kt) {
            Frag16 a[4];
            #pragma unroll
            for (int mt = 0; mt < 4; ++mt) {
                #pragma unroll
                for (int vv = 0; vv < 8; ++vv) {
                    int k0 = kt * 32 + half * 16 + vv * 2;
                    a[mt].u32[vv] =
                        *(const unsigned int*)&As[mt * 16 + l15][k0];
                }
            }
            Frag16 bb[2];
            #pragma unroll
            for (int ntl = 0; ntl < 2; ++ntl) {
                const unsigned int* src =
                    (const unsigned int*)&Bs[kt][wave * 2 + ntl][0];
                #pragma unroll
                for (int j = 0; j < 8; ++j) bb[ntl].u32[j] = src[lane * 8 + j];
            }
            #pragma unroll
            for (int mt = 0; mt < 4; ++mt)
                #pragma unroll
                for (int ntl = 0; ntl < 2; ++ntl)
                    acc[mt][ntl] = __builtin_amdgcn_wmma_f32_16x16x32_bf16(
                        false, a[mt].v, false, bb[ntl].v,
                        (short)0, acc[mt][ntl], false, false);
        }
    }

    // ---- fused bias + store fp32 ----
    #pragma unroll
    for (int mt = 0; mt < 4; ++mt)
        #pragma unroll
        for (int ntl = 0; ntl < 2; ++ntl) {
            int   ncol = (wave * 2 + ntl) * 16 + l15;
            float bv   = bias[ncol];
            #pragma unroll
            for (int vv = 0; vv < 8; ++vv) {
                int row = mt * 16 + vv + half * 8;
                out[(size_t)(b0 + row) * 256 + ncol] = acc[mt][ntl][vv] + bv;
            }
        }
}

// ---------------------------------------------------------------------------
// Recurrent scan:  for t in 0..511: h = tanh(xw[t] + h @ U)
// Batch rows are independent under h@U -> 4 blocks x 16 batch rows, no
// cross-block sync. 512 threads = 16 waves; wave w owns n-tile w.
// U held entirely in registers as bf16 B-fragments (8 frags x 8 VGPRs);
// h double-buffered in LDS as bf16. One split barrier per step; the next
// step's xw tile loads are issued in the signal->wait window so their
// global latency overlaps the WMMA chain instead of serializing with it.
// ---------------------------------------------------------------------------
template <bool STORE_SEQ>
__global__ __launch_bounds__(512) void rnn_scan_kernel(
    const float* __restrict__ xw, const float* __restrict__ U,
    unsigned short* __restrict__ seq_bf, float* __restrict__ h_last)
{
    __shared__ unsigned short hbuf[2][16][256];   // 16 KB double buffer

    const int tid  = threadIdx.x;
    const int lane = tid & 31;
    const int nt   = tid >> 5;          // wave id == n-tile
    const int half = lane >> 4;
    const int l15  = lane & 15;
    const int b0   = blockIdx.x * 16;   // batch-row base
    const int ncol = nt * 16 + l15;

    // ---- preload U as register-resident B fragments (reused 512 steps) ----
    Frag16 bU[8];
    #pragma unroll
    for (int kt = 0; kt < 8; ++kt)
        #pragma unroll
        for (int e = 0; e < 16; ++e) {
            int vv = e >> 1, odd = e & 1;
            int k  = kt * 32 + half * 16 + vv * 2 + odd;
            bU[kt].u16[e] = f2bf_bits(U[(size_t)k * 256 + ncol]);
        }

    // ---- h0 = 0 ----
    for (int i = tid; i < 16 * 256; i += 512) (&hbuf[0][0][0])[i] = 0;

    // ---- prologue: preload xw[0] tile; publish h0 ----
    v8f cin;
    {
        const float* cp = xw + (size_t)b0 * 256 + ncol;
        #pragma unroll
        for (int vv = 0; vv < 8; ++vv)
            cin[vv] = cp[(size_t)(vv + half * 8) * 256];
    }
    WG_BARRIER_SIGNAL();                // dscnt 0 (h0 stores) then arrive

    v8f c = {};
    for (int t = 0; t < 512; ++t) {
        const int src = t & 1;
        const int dst = src ^ 1;

        // issue next step's C loads NOW (pure global; overlaps barrier+WMMA)
        v8f cnext;
        {
            const int    tn = (t + 1) & 511;
            const float* cp = xw + ((size_t)tn * 64 + b0) * 256 + ncol;
            #pragma unroll
            for (int vv = 0; vv < 8; ++vv)
                cnext[vv] = cp[(size_t)(vv + half * 8) * 256];
        }

        WG_BARRIER_WAIT();              // hbuf[src] ready

        // h_prev @ U : 8 x v_wmma_f32_16x16x32_bf16, C initialized with xw[t]
        v8f cacc = cin;
        #pragma unroll
        for (int kt = 0; kt < 8; ++kt) {
            Frag16 a;
            #pragma unroll
            for (int vv = 0; vv < 8; ++vv) {
                int k0 = kt * 32 + half * 16 + vv * 2;
                a.u32[vv] = *(const unsigned int*)&hbuf[src][l15][k0];
            }
            cacc = __builtin_amdgcn_wmma_f32_16x16x32_bf16(
                false, a.v, false, bU[kt].v, (short)0, cacc, false, false);
        }

        #pragma unroll
        for (int vv = 0; vv < 8; ++vv) cacc[vv] = fast_tanh(cacc[vv]);

        // publish h_new (bf16) for next step / layer-2 A matrix
        #pragma unroll
        for (int vv = 0; vv < 8; ++vv) {
            int            row = vv + half * 8;
            unsigned short hb  = f2bf_bits(cacc[vv]);
            hbuf[dst][row][ncol] = hb;
            if (STORE_SEQ)
                seq_bf[((size_t)t * 64 + b0 + row) * 256 + ncol] = hb;
        }

        WG_BARRIER_SIGNAL();            // dscnt 0 (reads+writes retired), arrive
        c   = cacc;
        cin = cnext;                    // loadcnt wait lands here
    }

    if (!STORE_SEQ) {   // layer 2: only the final hidden state, full fp32
        #pragma unroll
        for (int vv = 0; vv < 8; ++vv)
            h_last[(size_t)(b0 + vv + half * 8) * 256 + ncol] = c[vv];
    }
}

// ---------------------------------------------------------------------------
// logits = h2 @ Wd + bd ; softmax over 2 classes. 64 rows -> 64 threads.
// ---------------------------------------------------------------------------
__global__ __launch_bounds__(64) void dense_softmax_kernel(
    const float* __restrict__ h2, const float* __restrict__ Wd,
    const float* __restrict__ bd, float* __restrict__ out)
{
    int   b  = threadIdx.x;
    float l0 = bd[0], l1 = bd[1];
    for (int k = 0; k < 256; ++k) {
        float h = h2[b * 256 + k];
        l0 = fmaf(h, Wd[k * 2 + 0], l0);
        l1 = fmaf(h, Wd[k * 2 + 1], l1);
    }
    float m  = fmaxf(l0, l1);
    float e0 = __expf(l0 - m), e1 = __expf(l1 - m);
    float s  = e0 + e1;
    out[b * 2 + 0] = e0 / s;
    out[b * 2 + 1] = e1 / s;
}

// ---------------------------------------------------------------------------
extern "C" void kernel_launch(void* const* d_in, const int* in_sizes, int n_in,
                              void* d_out, int out_size, void* d_ws, size_t ws_size,
                              hipStream_t stream) {
    (void)in_sizes; (void)n_in; (void)out_size; (void)ws_size;
    const int*   tokens = (const int*)  d_in[0];
    const float* emb    = (const float*)d_in[1];
    const float* W1     = (const float*)d_in[2];
    const float* U1     = (const float*)d_in[3];
    const float* b1     = (const float*)d_in[4];
    const float* W2     = (const float*)d_in[5];
    const float* U2     = (const float*)d_in[6];
    const float* b2     = (const float*)d_in[7];
    const float* Wd     = (const float*)d_in[8];
    const float* bd     = (const float*)d_in[9];
    float*       out    = (float*)d_out;

    char* ws = (char*)d_ws;
    // [0, 32MB)   : xw  (fp32, [512*64, 256]) -- reused for layer 1 and 2
    // [32, 48MB)  : h1 sequence (bf16, [512*64, 256])
    // [48MB, +64K): h2  (fp32, [64, 256])
    float*          xw    = (float*)ws;
    unsigned short* h1seq = (unsigned short*)(ws + (size_t)32 * 1024 * 1024);
    float*          h2    = (float*)(ws + (size_t)48 * 1024 * 1024);

    // 1) embed-gather + x@W1 + b1  (M=32768, K=512, N=256)
    proj_kernel<true><<<512, 256, 0, stream>>>(tokens, emb, nullptr, W1, b1, xw, 512);
    // 2) scan layer 1 (stores bf16 sequence)
    rnn_scan_kernel<true><<<4, 512, 0, stream>>>(xw, U1, h1seq, nullptr);
    // 3) h1@W2 + b2  (M=32768, K=256, N=256) -- overwrites dead xw1
    proj_kernel<false><<<512, 256, 0, stream>>>(nullptr, nullptr, h1seq, W2, b2, xw, 256);
    // 4) scan layer 2 (final hidden state only)
    rnn_scan_kernel<false><<<4, 512, 0, stream>>>(xw, U2, nullptr, h2);
    // 5) dense + softmax
    dense_softmax_kernel<<<1, 64, 0, stream>>>(h2, Wd, bd, out);
}
